// MultiHeadAttention_45990509805977
// MI455X (gfx1250) — compile-verified, matrix operand order checked by reading, not run on hardware
//
#include <hip/hip_runtime.h>

// ---------------- problem constants ----------------
#define BATCH 4
#define CDIM  1024
#define NH    16
#define HD    64          // head dim
#define WD    1500        // sequence (width)
#define WP    1504        // padded to multiple of 32

typedef __bf16 bf16_t;
typedef __attribute__((ext_vector_type(16))) __bf16 v16bf;
typedef __attribute__((ext_vector_type(8)))  __bf16 v8bf;
typedef __attribute__((ext_vector_type(8)))  float  v8f;

// ---------------------------------------------------------------------------
// Fragment loader for V_WMMA_F32_16X16X32_BF16 per CDNA5 ISA 7.12.2:
// A (16x32, MxK): lane L holds row M = L&15; lanes<16 carry K={0..7,16..23},
// lanes>=16 carry K={8..15,24..31}; each run of 8 bf16 is contiguous (16B).
// B is loaded from an n-major (Bt[n][k]) matrix with the mirrored layout
// (lane -> N, same K pattern), so both fragments are two 16B loads.
// ---------------------------------------------------------------------------
__device__ inline v16bf load_frag(const bf16_t* mat, int ld, int row0, int kb) {
    int lane  = threadIdx.x & 31;
    int m     = row0 + (lane & 15);
    int kbase = kb + ((lane & 16) ? 8 : 0);
    const bf16_t* p = mat + (size_t)m * ld + kbase;
    v8bf lo = *(const v8bf*)(p);
    v8bf hi = *(const v8bf*)(p + 16);
    v16bf r;
#pragma unroll
    for (int i = 0; i < 8; ++i) { r[i] = lo[i]; r[i + 8] = hi[i]; }
    return r;
}

__device__ inline v8f wmma_bf16(v16bf a, v16bf b, v8f c) {
    // (neg_a, A, neg_b, B, c_mod, C, reuse_a, reuse_b)
    return __builtin_amdgcn_wmma_f32_16x16x32_bf16(false, a, false, b,
                                                   (short)0, c, false, false);
}

// ---------------- prep: f32 -> bf16 weight convert ----------------
__global__ void cvt_bf16_kernel(const float* __restrict__ src,
                                bf16_t* __restrict__ dst, int n) {
    int i = blockIdx.x * blockDim.x + threadIdx.x;
    if (i < n) dst[i] = (bf16_t)src[i];
}

// ---------------- prep: x [B][C][W] -> xT bf16 [B][WP][C], zero padded ------
__global__ void transpose_x_kernel(const float* __restrict__ x,
                                   bf16_t* __restrict__ xT) {
    int i = blockIdx.x * blockDim.x + threadIdx.x;
    if (i >= BATCH * WP * CDIM) return;
    int c = i & (CDIM - 1);
    int w = (i / CDIM) % WP;
    int b = i / (CDIM * WP);
    float v = (w < WD) ? x[((size_t)b * CDIM + c) * WD + w] : 0.0f;
    xT[i] = (bf16_t)v;
}

// ---------------- QKV projection GEMM ---------------------------------------
// y[o,p] = sum_c W[o,c] * x[c,p]; A = W (bf16, row-major o x c),
// Bt = xT (bf16, p-major). One wave computes a 32x32 output tile.
// which: 0 -> q (bias+scale, layout [b][h][w][d])
//        1 -> k (layout [b][h][w][d])
//        2 -> v (bias, layout [b][c][w])
__global__ __launch_bounds__(32) void qkv_gemm_kernel(
    const bf16_t* __restrict__ wq, const bf16_t* __restrict__ wk,
    const bf16_t* __restrict__ wv, const bf16_t* __restrict__ xT,
    const float* __restrict__ bq, const float* __restrict__ bv,
    bf16_t* __restrict__ qws, bf16_t* __restrict__ kws,
    bf16_t* __restrict__ vws) {
    const int n0 = blockIdx.x * 32;
    const int m0 = blockIdx.y * 32;
    const int b  = blockIdx.z / 3;
    const int which = blockIdx.z % 3;
    const bf16_t* Wm = (which == 0) ? wq : (which == 1) ? wk : wv;
    const bf16_t* xb = xT + (size_t)b * WP * CDIM;

    v8f acc[2][2] = {};
    for (int kb = 0; kb < CDIM; kb += 32) {
        if (kb + 32 < CDIM) {  // gfx1250 global_prefetch_b8 hints
            __builtin_prefetch(Wm + (size_t)m0 * CDIM + kb + 32, 0, 1);
            __builtin_prefetch(xb + (size_t)n0 * CDIM + kb + 32, 0, 1);
        }
        v16bf a0 = load_frag(Wm, CDIM, m0,      kb);
        v16bf a1 = load_frag(Wm, CDIM, m0 + 16, kb);
        v16bf b0 = load_frag(xb, CDIM, n0,      kb);
        v16bf b1 = load_frag(xb, CDIM, n0 + 16, kb);
        acc[0][0] = wmma_bf16(a0, b0, acc[0][0]);
        acc[0][1] = wmma_bf16(a0, b1, acc[0][1]);
        acc[1][0] = wmma_bf16(a1, b0, acc[1][0]);
        acc[1][1] = wmma_bf16(a1, b1, acc[1][1]);
    }

    const int lane  = threadIdx.x & 31;
    const int ncol  = lane & 15;
    const int rbase = (lane & 16) ? 8 : 0;
    const float scale = 0.125f;  // 64^-0.5
#pragma unroll
    for (int i = 0; i < 2; ++i)
#pragma unroll
        for (int j = 0; j < 2; ++j)
#pragma unroll
            for (int r = 0; r < 8; ++r) {
                int o = m0 + i * 16 + rbase + r;  // channel
                int p = n0 + j * 16 + ncol;       // position (< WP)
                float v = acc[i][j][r];
                if (which == 0) {
                    v = (v + bq[o]) * scale;
                    qws[(((size_t)(b * NH + (o >> 6)) * WP + p) << 6) + (o & 63)] =
                        (bf16_t)v;
                } else if (which == 1) {
                    kws[(((size_t)(b * NH + (o >> 6)) * WP + p) << 6) + (o & 63)] =
                        (bf16_t)v;
                } else {
                    v += bv[o];
                    vws[((size_t)(b * CDIM + o)) * WP + p] = (bf16_t)v;
                }
            }
}

// ---------------- flash attention: one wave per 16-query tile ---------------
__global__ __launch_bounds__(32) void attn_kernel(
    const bf16_t* __restrict__ qws, const bf16_t* __restrict__ kws,
    const bf16_t* __restrict__ vws, bf16_t* __restrict__ ows) {
    const int q0 = blockIdx.x * 16;
    const int bh = blockIdx.y;  // b*16 + h
    const int b  = bh >> 4;
    const int h  = bh & 15;
    const bf16_t* qh = qws + (size_t)bh * WP * HD;
    const bf16_t* kh = kws + (size_t)bh * WP * HD;
    const bf16_t* vh = vws + ((size_t)b * CDIM + h * HD) * WP;  // [d][WP]

    __shared__ bf16_t plds[16 * 32];
    const int lane  = threadIdx.x & 31;
    const int ncol  = lane & 15;
    const int rbase = (lane & 16) ? 8 : 0;

    // Q fragments (row-major [w][d], two K=32 chunks cover d=64)
    v16bf aq0 = load_frag(qh, HD, q0, 0);
    v16bf aq1 = load_frag(qh, HD, q0, 32);

    v8f o0 = {}, o1 = {}, o2 = {}, o3 = {};
    float mrow[8], lrow[8];
#pragma unroll
    for (int r = 0; r < 8; ++r) { mrow[r] = -3.0e38f; lrow[r] = 0.0f; }

    for (int jb = 0; jb < WP; jb += 32) {
        // scores for 32 keys: two 16x16 tiles, K=64 via two WMMAs each
        v16bf bk0 = load_frag(kh, HD, jb,      0);
        v16bf bk1 = load_frag(kh, HD, jb,      32);
        v16bf bk2 = load_frag(kh, HD, jb + 16, 0);
        v16bf bk3 = load_frag(kh, HD, jb + 16, 32);
        v8f s0 = {}, s1 = {};
        s0 = wmma_bf16(aq0, bk0, s0);
        s0 = wmma_bf16(aq1, bk1, s0);
        s1 = wmma_bf16(aq0, bk2, s1);
        s1 = wmma_bf16(aq1, bk3, s1);

        const bool mask0 = (jb + ncol) >= WD;
        const bool mask1 = (jb + 16 + ncol) >= WD;

        float p0[8], p1[8], alpha[8];
#pragma unroll
        for (int r = 0; r < 8; ++r) {
            float v0 = mask0 ? -3.0e38f : s0[r];
            float v1 = mask1 ? -3.0e38f : s1[r];
            float mx = fmaxf(v0, v1);
            mx = fmaxf(mx, __shfl_xor(mx, 1, 32));
            mx = fmaxf(mx, __shfl_xor(mx, 2, 32));
            mx = fmaxf(mx, __shfl_xor(mx, 4, 32));
            mx = fmaxf(mx, __shfl_xor(mx, 8, 32));
            float mn = fmaxf(mrow[r], mx);
            float e0 = __expf(v0 - mn);
            float e1 = __expf(v1 - mn);
            float rs = e0 + e1;
            rs += __shfl_xor(rs, 1, 32);
            rs += __shfl_xor(rs, 2, 32);
            rs += __shfl_xor(rs, 4, 32);
            rs += __shfl_xor(rs, 8, 32);
            float al = __expf(mrow[r] - mn);
            lrow[r]  = lrow[r] * al + rs;
            mrow[r]  = mn;
            alpha[r] = al;
            p0[r] = e0;
            p1[r] = e1;
        }
#pragma unroll
        for (int r = 0; r < 8; ++r) {
            o0[r] *= alpha[r]; o1[r] *= alpha[r];
            o2[r] *= alpha[r]; o3[r] *= alpha[r];
        }
        // C-layout P -> LDS -> A fragment (cross-lane transpose)
        __syncthreads();
#pragma unroll
        for (int r = 0; r < 8; ++r) {
            plds[(rbase + r) * 32 + ncol]      = (bf16_t)p0[r];
            plds[(rbase + r) * 32 + 16 + ncol] = (bf16_t)p1[r];
        }
        __syncthreads();
        v16bf pa = load_frag(plds, 32, 0, 0);

        // O += P * V ; V is [d][WP] so B fragments are contiguous in j
        v16bf bv0 = load_frag(vh, WP, 0,  jb);
        v16bf bv1 = load_frag(vh, WP, 16, jb);
        v16bf bv2 = load_frag(vh, WP, 32, jb);
        v16bf bv3 = load_frag(vh, WP, 48, jb);
        o0 = wmma_bf16(pa, bv0, o0);
        o1 = wmma_bf16(pa, bv1, o1);
        o2 = wmma_bf16(pa, bv2, o2);
        o3 = wmma_bf16(pa, bv3, o3);
    }

#pragma unroll
    for (int r = 0; r < 8; ++r) {
        float inv = 1.0f / lrow[r];
        o0[r] *= inv; o1[r] *= inv; o2[r] *= inv; o3[r] *= inv;
    }
    // store O as [b][w][c] bf16 for the n-major output-GEMM fragment path
#pragma unroll
    for (int r = 0; r < 8; ++r) {
        int w = q0 + rbase + r;
        size_t base = ((size_t)b * WP + w) * CDIM + h * HD + ncol;
        ows[base +  0] = (bf16_t)o0[r];
        ows[base + 16] = (bf16_t)o1[r];
        ows[base + 32] = (bf16_t)o2[r];
        ows[base + 48] = (bf16_t)o3[r];
    }
}

// ---------------- output projection GEMM + bias -----------------------------
__global__ __launch_bounds__(32) void out_gemm_kernel(
    const bf16_t* __restrict__ wo, const bf16_t* __restrict__ ows,
    const float* __restrict__ bo, float* __restrict__ out) {
    const int n0 = blockIdx.x * 32;
    const int m0 = blockIdx.y * 32;
    const int b  = blockIdx.z;
    const bf16_t* ob = ows + (size_t)b * WP * CDIM;

    v8f acc[2][2] = {};
    for (int kb = 0; kb < CDIM; kb += 32) {
        if (kb + 32 < CDIM) {
            __builtin_prefetch(wo + (size_t)m0 * CDIM + kb + 32, 0, 1);
            __builtin_prefetch(ob + (size_t)n0 * CDIM + kb + 32, 0, 1);
        }
        v16bf a0 = load_frag(wo, CDIM, m0,      kb);
        v16bf a1 = load_frag(wo, CDIM, m0 + 16, kb);
        v16bf b0 = load_frag(ob, CDIM, n0,      kb);
        v16bf b1 = load_frag(ob, CDIM, n0 + 16, kb);
        acc[0][0] = wmma_bf16(a0, b0, acc[0][0]);
        acc[0][1] = wmma_bf16(a0, b1, acc[0][1]);
        acc[1][0] = wmma_bf16(a1, b0, acc[1][0]);
        acc[1][1] = wmma_bf16(a1, b1, acc[1][1]);
    }

    const int lane  = threadIdx.x & 31;
    const int ncol  = lane & 15;
    const int rbase = (lane & 16) ? 8 : 0;
#pragma unroll
    for (int i = 0; i < 2; ++i)
#pragma unroll
        for (int j = 0; j < 2; ++j)
#pragma unroll
            for (int r = 0; r < 8; ++r) {
                int o = m0 + i * 16 + rbase + r;
                int p = n0 + j * 16 + ncol;
                if (p < WD)
                    out[((size_t)b * CDIM + o) * WD + p] = acc[i][j][r] + bo[o];
            }
}

// ---------------- host side -------------------------------------------------
extern "C" void kernel_launch(void* const* d_in, const int* in_sizes, int n_in,
                              void* d_out, int out_size, void* d_ws,
                              size_t ws_size, hipStream_t stream) {
    const float* x  = (const float*)d_in[0];
    const float* Wq = (const float*)d_in[1];
    const float* bq = (const float*)d_in[2];
    const float* Wk = (const float*)d_in[3];
    const float* Wv = (const float*)d_in[4];
    const float* bv = (const float*)d_in[5];
    const float* Wo = (const float*)d_in[6];
    const float* bo = (const float*)d_in[7];
    float* out = (float*)d_out;

    // workspace carve (256B aligned)
    size_t off = 0;
    auto carve = [&](size_t bytes) {
        void* p = (char*)d_ws + off;
        off += (bytes + 255) & ~(size_t)255;
        return p;
    };
    const size_t wbytes = (size_t)CDIM * CDIM * sizeof(bf16_t);
    bf16_t* wqb = (bf16_t*)carve(wbytes);
    bf16_t* wkb = (bf16_t*)carve(wbytes);
    bf16_t* wvb = (bf16_t*)carve(wbytes);
    bf16_t* wob = (bf16_t*)carve(wbytes);
    bf16_t* xT  = (bf16_t*)carve((size_t)BATCH * WP * CDIM * sizeof(bf16_t));
    bf16_t* qws = (bf16_t*)carve((size_t)BATCH * NH * WP * HD * sizeof(bf16_t));
    bf16_t* kws = (bf16_t*)carve((size_t)BATCH * NH * WP * HD * sizeof(bf16_t));
    bf16_t* vws = (bf16_t*)carve((size_t)BATCH * CDIM * WP * sizeof(bf16_t));
    bf16_t* ows = (bf16_t*)carve((size_t)BATCH * WP * CDIM * sizeof(bf16_t));
    (void)ws_size; (void)n_in; (void)in_sizes; (void)out_size;

    // 1) weights -> bf16
    {
        int n = CDIM * CDIM;
        int g = (n + 255) / 256;
        cvt_bf16_kernel<<<g, 256, 0, stream>>>(Wq, wqb, n);
        cvt_bf16_kernel<<<g, 256, 0, stream>>>(Wk, wkb, n);
        cvt_bf16_kernel<<<g, 256, 0, stream>>>(Wv, wvb, n);
        cvt_bf16_kernel<<<g, 256, 0, stream>>>(Wo, wob, n);
    }
    // 2) x -> xT bf16 [B][WP][C] (zero padded rows)
    {
        int n = BATCH * WP * CDIM;
        transpose_x_kernel<<<(n + 255) / 256, 256, 0, stream>>>(x, xT);
    }
    // 3) fused QKV projection
    qkv_gemm_kernel<<<dim3(WP / 32, CDIM / 32, BATCH * 3), 32, 0, stream>>>(
        wqb, wkb, wvb, xT, bq, bv, qws, kws, vws);
    // 4) flash attention
    attn_kernel<<<dim3(WP / 16, BATCH * NH), 32, 0, stream>>>(qws, kws, vws, ows);
    // 5) output projection + bias
    out_gemm_kernel<<<dim3(WP / 32, CDIM / 32, BATCH), 32, 0, stream>>>(
        wob, ows, bo, out);
}
